// MultiHeadAttentionMemoryCompressed_5257039970387
// MI455X (gfx1250) — matrix operand order repaired and checked by reading, not compiled, not run on hardware
//
#include <hip/hip_runtime.h>
#include <hip/hip_bf16.h>

// ---------------------------------------------------------------------------
// MultiHeadAttentionMemoryCompressed for MI455X (gfx1250), wave32 + WMMA bf16.
// Shapes: Lq=2048, B=4, Lk=1024, D=1024, DK=512, H=16, Dh=64.
// Pipeline: f32->bf16 converts, weight transposes, 4x WMMA GEMMs (software-
// pipelined k-loop), fused WMMA attention (scores, masked softmax, coverage,
// attn@V with double-buffered fragments).
// ---------------------------------------------------------------------------

typedef __attribute__((ext_vector_type(16))) __bf16 v16bf;
typedef __attribute__((ext_vector_type(8)))  __bf16 v8bf;
typedef __attribute__((ext_vector_type(8)))  float  v8f;

constexpr int NB  = 4;     // batch
constexpr int NH  = 16;    // heads
constexpr int LQ  = 2048;
constexpr int LKk = 1024;
constexpr int DM  = 1024;  // d_model
constexpr int DKc = 512;
constexpr int DH  = 64;    // head dim

// ---- WMMA fragment loaders (ISA 7.12.2 layouts, wave32) --------------------
// A-matrix 16x32 bf16: lane l holds row (l&15); elements 0..7 = K off..off+7,
// elements 8..15 = K 16+off..+7, off = (l>>4)*8. Two 16B contiguous loads.
__device__ __forceinline__ v16bf load_a_frag(const __bf16* A, int lda) {
  const int lane = threadIdx.x & 31;
  const int row  = lane & 15;
  const int off  = (lane >> 4) << 3;
  const __bf16* p = A + (size_t)row * lda + off;
  v8bf lo = *(const v8bf*)(p);
  v8bf hi = *(const v8bf*)(p + 16);
  v16bf r;
#pragma unroll
  for (int e = 0; e < 8; ++e) { r[e] = lo[e]; r[e + 8] = hi[e]; }
  return r;
}

// B-matrix 32x16 bf16 loaded from Bt (N x K row-major): lane l holds column
// (l&15); elements 0..7 = K off..off+7, 8..15 = K 16+off..+7, off=(l>>4)*8.
__device__ __forceinline__ v16bf load_b_frag(const __bf16* Bt, int ldb) {
  const int lane = threadIdx.x & 31;
  const int n    = lane & 15;
  const int koff = (lane >> 4) << 3;
  const __bf16* p = Bt + (size_t)n * ldb + koff;
  v8bf lo = *(const v8bf*)(p);
  v8bf hi = *(const v8bf*)(p + 16);
  v16bf r;
#pragma unroll
  for (int e = 0; e < 8; ++e) { r[e] = lo[e]; r[e + 8] = hi[e]; }
  return r;
}

__device__ __forceinline__ v8f wmma_bf16(v16bf a, v16bf b, v8f c) {
  return __builtin_amdgcn_wmma_f32_16x16x32_bf16(
      /*neg_a=*/false, a, /*neg_b=*/false, b,
      /*c_mod=*/(short)0, c, /*reuse_a=*/false, /*reuse_b=*/false);
}

// ---- elementwise converts --------------------------------------------------
__global__ void __launch_bounds__(256)
f32_to_bf16_k(const float* __restrict__ in, __bf16* __restrict__ out, int n) {
  int i = (blockIdx.x * 256 + threadIdx.x) * 4;
  if (i + 3 < n) {
    float4 v = *(const float4*)(in + i);
    out[i + 0] = (__bf16)v.x; out[i + 1] = (__bf16)v.y;
    out[i + 2] = (__bf16)v.z; out[i + 3] = (__bf16)v.w;
  } else {
    for (; i < n; ++i) out[i] = (__bf16)in[i];
  }
}

// W (K,N) f32 -> Wt (N,K) bf16
__global__ void __launch_bounds__(256)
transpose_bf16_k(const float* __restrict__ W, __bf16* __restrict__ Wt,
                 int K, int N) {
  int idx = blockIdx.x * 256 + threadIdx.x;
  if (idx >= K * N) return;
  int k = idx / N, n = idx - k * N;
  Wt[(size_t)n * K + k] = (__bf16)W[idx];
}

// ---- generic WMMA GEMM: C = A(MxK,bf16) * Bt^T (Bt is NxK bf16) ------------
// Each wave computes a 16x64 strip (1 m-tile x 4 n-tiles); k-loop is software
// pipelined (ping-pong fragment double-buffering, 2x32 per iteration).
// K must be a multiple of 64 (here 512 or 1024).
// mode 0: Q-proj  -> Qbf  [(b*H+h)*LQ + q]*64 + d      (bf16, scaled)
// mode 1: K-proj  -> Kbf  [(b*H+h)*LK + lk]*64 + d     (bf16)
// mode 2: V-proj  -> Vtbf [(b*H+h)*64 + d]*LK + lk     (bf16, transposed)
// mode 3: final   -> Cf   [m*N + n]                     (f32)
__global__ void __launch_bounds__(256)
gemm16x64_bf16(const __bf16* __restrict__ A, const __bf16* __restrict__ Bt,
               float* __restrict__ Cf, __bf16* __restrict__ Cb,
               int M, int N, int K, int mode, float scale) {
  const int stripsN = N >> 6;
  const int wave = blockIdx.x * (blockDim.x >> 5) + (threadIdx.x >> 5);
  if (wave >= (M >> 4) * stripsN) return;  // wave-uniform exit
  const int m0 = (wave / stripsN) << 4;
  const int n0 = (wave % stripsN) << 6;

  const __bf16* Ap = A + (size_t)m0 * K;
  const __bf16* Bp = Bt + (size_t)n0 * K;

  v8f acc[4] = {};

  // prologue: stage-0 fragments (k = 0)
  v16bf a0 = load_a_frag(Ap, K);
  v16bf b0[4];
#pragma unroll
  for (int j = 0; j < 4; ++j) b0[j] = load_b_frag(Bp + (size_t)(j << 4) * K, K);

  for (int k = 0; k < K; k += 64) {
    // issue stage-1 loads (k+32) before consuming stage-0
    v16bf a1 = load_a_frag(Ap + k + 32, K);
    v16bf b1[4];
#pragma unroll
    for (int j = 0; j < 4; ++j)
      b1[j] = load_b_frag(Bp + (size_t)(j << 4) * K + k + 32, K);

#pragma unroll
    for (int j = 0; j < 4; ++j) acc[j] = wmma_bf16(a0, b0[j], acc[j]);

    // issue stage-0 loads for k+64 before consuming stage-1
    if (k + 64 < K) {
      a0 = load_a_frag(Ap + k + 64, K);
#pragma unroll
      for (int j = 0; j < 4; ++j)
        b0[j] = load_b_frag(Bp + (size_t)(j << 4) * K + k + 64, K);
    }

#pragma unroll
    for (int j = 0; j < 4; ++j) acc[j] = wmma_bf16(a1, b1[j], acc[j]);
  }

  const int lane  = threadIdx.x & 31;
  const int rbase = (lane >> 4) << 3;  // D-tile: vgpr i -> row i + 8*(lane>>4)
  const int cl    = lane & 15;
#pragma unroll
  for (int j = 0; j < 4; ++j) {
#pragma unroll
    for (int i = 0; i < 8; ++i) {
      const int gm = m0 + rbase + i;
      const int gn = n0 + (j << 4) + cl;
      const float v = acc[j][i] * scale;
      if (mode == 3) {
        Cf[(size_t)gm * N + gn] = v;
      } else {
        const int h = gn >> 6, d = gn & 63;
        const int rowq = gm >> 2, bb = gm & 3;  // memory row m = seq*B + b
        if (mode == 0)
          Cb[(((size_t)(bb * NH + h)) * LQ + rowq) * DH + d] = (__bf16)v;
        else if (mode == 1)
          Cb[(((size_t)(bb * NH + h)) * LKk + rowq) * DH + d] = (__bf16)v;
        else
          Cb[(((size_t)(bb * NH + h)) * DH + d) * LKk + rowq] = (__bf16)v;
      }
    }
  }
}

// ---- fused attention -------------------------------------------------------
// grid: B * (LQ/16) blocks, 256 threads (8 waves). Wave w owns k-columns
// [w*128, w*128+128). Loop over all 16 heads so coverage accumulates in regs.
__global__ void __launch_bounds__(256)
attn_fused(const __bf16* __restrict__ Qbf, const __bf16* __restrict__ Kbf,
           const __bf16* __restrict__ Vtbf, const unsigned char* __restrict__ mask,
           __bf16* __restrict__ attn_bf, float* __restrict__ cov) {
  __shared__ __bf16 Psh[16 * LKk];   // 32 KB: probabilities, bf16 row-major
  __shared__ float  Osh[16 * DH];    //  4 KB: cross-wave O accumulator
  __shared__ float  redmax[8][16];
  __shared__ float  redsum[8][16];

  const int b    = blockIdx.x >> 7;     // LQ/16 = 128 q-tiles
  const int q0   = (blockIdx.x & 127) << 4;
  const int w    = threadIdx.x >> 5;
  const int lane = threadIdx.x & 31;
  const int hf   = lane >> 4;
  const int cl   = lane & 15;
  const int k0w  = w << 7;              // this wave's 128-col base

  for (int e = threadIdx.x; e < 16 * DH; e += 256) Osh[e] = 0.f;

  // Pack mask (head-invariant) into 64 bits/lane: bit kt*8+i for score tile
  // element (vgpr i, tile kt) = row i+8*hf, col k0w+kt*16+cl.
  unsigned long long mbits = 0ull;
  {
    const unsigned char* mrow = mask + ((size_t)b * LQ + q0) * LKk;
#pragma unroll
    for (int kt = 0; kt < 8; ++kt)
#pragma unroll
      for (int i = 0; i < 8; ++i) {
        const int r = (hf << 3) + i;
        const int c = k0w + (kt << 4) + cl;
        mbits |= ((unsigned long long)(mrow[(size_t)r * LKk + c] != 0))
                 << (kt * 8 + i);
      }
  }

  float covacc[8][8] = {};  // coverage accumulator, matches score D-layout

  for (int h = 0; h < NH; ++h) {
    const __bf16* Qbh = Qbf + (((size_t)(b * NH + h)) * LQ + q0) * DH;
    const __bf16* Kbh = Kbf + ((size_t)(b * NH + h)) * LKk * DH;
    const __bf16* Vbh = Vtbf + ((size_t)(b * NH + h)) * DH * LKk;

    const v16bf qa0 = load_a_frag(Qbh, DH);        // d = 0..31
    const v16bf qa1 = load_a_frag(Qbh + 32, DH);   // d = 32..63

    // scores: batch all 8 K-fragment loads per d-chunk, then 8 independent
    // WMMA chains (lets the scheduler overlap waits with math).
    v8f s[8];
    v16bf kb[8];
#pragma unroll
    for (int kt = 0; kt < 8; ++kt)
      kb[kt] = load_b_frag(Kbh + (size_t)(k0w + (kt << 4)) * DH, DH);
#pragma unroll
    for (int kt = 0; kt < 8; ++kt) {
      v8f z = {};
      s[kt] = wmma_bf16(qa0, kb[kt], z);
    }
#pragma unroll
    for (int kt = 0; kt < 8; ++kt)
      kb[kt] = load_b_frag(Kbh + (size_t)(k0w + (kt << 4)) * DH + 32, DH);
#pragma unroll
    for (int kt = 0; kt < 8; ++kt) s[kt] = wmma_bf16(qa1, kb[kt], s[kt]);

    // mask + per-lane row-max partials
    float m[8];
#pragma unroll
    for (int i = 0; i < 8; ++i) m[i] = -3.0e38f;
#pragma unroll
    for (int kt = 0; kt < 8; ++kt)
#pragma unroll
      for (int i = 0; i < 8; ++i) {
        float v = s[kt][i];
        if (mbits & (1ull << (kt * 8 + i))) v = -3.0e38f;
        s[kt][i] = v;
        m[i] = fmaxf(m[i], v);
      }
    // reduce across the 16 lanes of each half (cols), then across 8 waves
#pragma unroll
    for (int off = 1; off < 16; off <<= 1)
#pragma unroll
      for (int i = 0; i < 8; ++i) m[i] = fmaxf(m[i], __shfl_xor(m[i], off, 32));
    if (cl == 0) {
#pragma unroll
      for (int i = 0; i < 8; ++i) redmax[w][(hf << 3) + i] = m[i];
    }
    __syncthreads();
    float rmax[8];
#pragma unroll
    for (int i = 0; i < 8; ++i) {
      const int r = (hf << 3) + i;
      float mm = redmax[0][r];
#pragma unroll
      for (int ww = 1; ww < 8; ++ww) mm = fmaxf(mm, redmax[ww][r]);
      rmax[i] = mm;
    }

    // exp + row-sum
    float su[8] = {};
#pragma unroll
    for (int kt = 0; kt < 8; ++kt)
#pragma unroll
      for (int i = 0; i < 8; ++i) {
        const float e = __expf(s[kt][i] - rmax[i]);
        s[kt][i] = e;
        su[i] += e;
      }
#pragma unroll
    for (int off = 1; off < 16; off <<= 1)
#pragma unroll
      for (int i = 0; i < 8; ++i) su[i] += __shfl_xor(su[i], off, 32);
    if (cl == 0) {
#pragma unroll
      for (int i = 0; i < 8; ++i) redsum[w][(hf << 3) + i] = su[i];
    }
    __syncthreads();
    float rinv[8];
#pragma unroll
    for (int i = 0; i < 8; ++i) {
      const int r = (hf << 3) + i;
      float ss = redsum[0][r];
#pragma unroll
      for (int ww = 1; ww < 8; ++ww) ss += redsum[ww][r];
      rinv[i] = 1.0f / ss;
    }

    // normalize -> coverage accumulate -> stash P (bf16) in LDS row-major
#pragma unroll
    for (int kt = 0; kt < 8; ++kt)
#pragma unroll
      for (int i = 0; i < 8; ++i) {
        const float p = s[kt][i] * rinv[i];
        covacc[kt][i] += p * (1.0f / (float)NH);
        const int r = (hf << 3) + i;
        const int c = k0w + (kt << 4) + cl;
        Psh[r * LKk + c] = (__bf16)p;
      }

    // O_partial = P[:, k0w:+128] @ V[k0w:+128, :], double-buffered fragments
    v8f oacc[4] = {};
    v16bf pa = load_a_frag(&Psh[k0w], LKk);
    v16bf vb[4];
#pragma unroll
    for (int nt = 0; nt < 4; ++nt)
      vb[nt] = load_b_frag(Vbh + (size_t)(nt << 4) * LKk + k0w, LKk);
#pragma unroll
    for (int ks = 0; ks < 4; ++ks) {
      v16bf pan;
      v16bf vbn[4];
      if (ks < 3) {  // prefetch next k-chunk before consuming current
        pan = load_a_frag(&Psh[k0w + ((ks + 1) << 5)], LKk);
#pragma unroll
        for (int nt = 0; nt < 4; ++nt)
          vbn[nt] = load_b_frag(
              Vbh + (size_t)(nt << 4) * LKk + k0w + ((ks + 1) << 5), LKk);
      }
#pragma unroll
      for (int nt = 0; nt < 4; ++nt) oacc[nt] = wmma_bf16(pa, vb[nt], oacc[nt]);
      if (ks < 3) {
        pa = pan;
#pragma unroll
        for (int nt = 0; nt < 4; ++nt) vb[nt] = vbn[nt];
      }
    }

    // cross-wave reduce via ds_add_f32
#pragma unroll
    for (int nt = 0; nt < 4; ++nt)
#pragma unroll
      for (int i = 0; i < 8; ++i) {
        const int r = (hf << 3) + i;
        const int d = (nt << 4) + cl;
        atomicAdd(&Osh[r * DH + d], oacc[nt][i]);
      }
    __syncthreads();
    // emit O tile (bf16, layout (Lq,B,H*Dh)) and re-zero accumulator
    for (int e = threadIdx.x; e < 16 * DH; e += 256) {
      const int r = e >> 6, d = e & 63;
      attn_bf[(((size_t)(q0 + r)) * NB + b) * DM + h * DH + d] = (__bf16)Osh[e];
      Osh[e] = 0.f;
    }
    __syncthreads();
  }

  // coverage (B, Lq, Lk) f32
#pragma unroll
  for (int kt = 0; kt < 8; ++kt)
#pragma unroll
    for (int i = 0; i < 8; ++i) {
      const int r = (hf << 3) + i;
      const int c = k0w + (kt << 4) + cl;
      cov[((size_t)b * LQ + q0 + r) * LKk + c] = covacc[kt][i];
    }
}

// ---------------------------------------------------------------------------
extern "C" void kernel_launch(void* const* d_in, const int* in_sizes, int n_in,
                              void* d_out, int out_size, void* d_ws, size_t ws_size,
                              hipStream_t stream) {
  const float* query = (const float*)d_in[0];  // (Lq,B,D)
  const float* key   = (const float*)d_in[1];  // (Lk,B,DK)
  // d_in[2] = value: unused (reference computes V from `key`, matching it)
  const unsigned char* mask = (const unsigned char*)d_in[3];  // (B,Lq,Lk) bool
  const float* Wq = (const float*)d_in[4];  // (D, D)
  const float* Wk = (const float*)d_in[5];  // (DK, D)
  const float* Wv = (const float*)d_in[6];  // (DK, D)
  const float* Wc = (const float*)d_in[7];  // (D, D)

  float* out_f = (float*)d_out;                         // (Lq,B,D)
  float* cov_f = out_f + (size_t)LQ * NB * DM;          // (B,Lq,Lk)

  // workspace carve-out (bf16 buffers), 256B aligned
  char* ws = (char*)d_ws;
  size_t off = 0;
  auto carve = [&](size_t bytes) -> void* {
    void* p = ws + off;
    off = (off + bytes + 255) & ~(size_t)255;
    return p;
  };
  const size_t Mq = (size_t)LQ * NB;   // 8192 query rows
  const size_t Mk = (size_t)LKk * NB;  // 4096 key rows
  __bf16* query_bf = (__bf16*)carve(Mq * DM * 2);
  __bf16* key_bf   = (__bf16*)carve(Mk * DKc * 2);
  __bf16* Wq_t     = (__bf16*)carve((size_t)DM * DM * 2);
  __bf16* Wk_t     = (__bf16*)carve((size_t)DM * DKc * 2);
  __bf16* Wv_t     = (__bf16*)carve((size_t)DM * DKc * 2);
  __bf16* Wc_t     = (__bf16*)carve((size_t)DM * DM * 2);
  __bf16* Qbf      = (__bf16*)carve((size_t)NB * NH * LQ * DH * 2);
  __bf16* Kbf      = (__bf16*)carve((size_t)NB * NH * LKk * DH * 2);
  __bf16* Vtbf     = (__bf16*)carve((size_t)NB * NH * DH * LKk * 2);
  __bf16* attn_bf  = (__bf16*)carve(Mq * DM * 2);

  // 1) converts
  {
    int n = (int)(Mq * DM);
    f32_to_bf16_k<<<(n / 4 + 255) / 256, 256, 0, stream>>>(query, query_bf, n);
  }
  {
    int n = (int)(Mk * DKc);
    f32_to_bf16_k<<<(n / 4 + 255) / 256, 256, 0, stream>>>(key, key_bf, n);
  }
  transpose_bf16_k<<<(DM * DM + 255) / 256, 256, 0, stream>>>(Wq, Wq_t, DM, DM);
  transpose_bf16_k<<<(DKc * DM + 255) / 256, 256, 0, stream>>>(Wk, Wk_t, DKc, DM);
  transpose_bf16_k<<<(DKc * DM + 255) / 256, 256, 0, stream>>>(Wv, Wv_t, DKc, DM);
  transpose_bf16_k<<<(DM * DM + 255) / 256, 256, 0, stream>>>(Wc, Wc_t, DM, DM);

  // 2) projections (scale Dh^-0.5 fused into Q)
  {
    int waves = ((int)Mq >> 4) * (DM >> 6);
    gemm16x64_bf16<<<(waves + 7) / 8, 256, 0, stream>>>(
        query_bf, Wq_t, nullptr, Qbf, (int)Mq, DM, DM, /*mode=*/0, 0.125f);
  }
  {
    int waves = ((int)Mk >> 4) * (DM >> 6);
    gemm16x64_bf16<<<(waves + 7) / 8, 256, 0, stream>>>(
        key_bf, Wk_t, nullptr, Kbf, (int)Mk, DM, DKc, /*mode=*/1, 1.0f);
    gemm16x64_bf16<<<(waves + 7) / 8, 256, 0, stream>>>(
        key_bf, Wv_t, nullptr, Vtbf, (int)Mk, DM, DKc, /*mode=*/2, 1.0f);
  }

  // 3) fused attention (+ coverage)
  attn_fused<<<NB * (LQ / 16), 256, 0, stream>>>(Qbf, Kbf, Vtbf, mask,
                                                 attn_bf, cov_f);

  // 4) output projection -> f32 d_out
  {
    int waves = ((int)Mq >> 4) * (DM >> 6);
    gemm16x64_bf16<<<(waves + 7) / 8, 256, 0, stream>>>(
        attn_bf, Wc_t, out_f, nullptr, (int)Mq, DM, DM, /*mode=*/3, 1.0f);
  }
}
